// GCNAutoEncoder_31370441130067
// MI455X (gfx1250) — matrix operand order repaired and checked by reading, not compiled, not run on hardware
//
#include <hip/hip_runtime.h>
#include <hip/hip_bf16.h>
#include <cstdint>

typedef __attribute__((ext_vector_type(16))) _Float16 v16h;
typedef __attribute__((ext_vector_type(8)))  float    v8f;
typedef __attribute__((ext_vector_type(4)))  float    v4f;

#define TPB 256

// ---------------- degree / normalization ----------------

__global__ void fill_f32(float* p, float v, int n) {
  int i = blockIdx.x * TPB + threadIdx.x;
  if (i < n) p[i] = v;
}

__global__ void deg_count(const long long* __restrict__ ei, float* deg, int E) {
  int e = blockIdx.x * TPB + threadIdx.x;
  if (e < E) atomicAdd(&deg[(int)ei[(size_t)E + e]], 1.0f);
}

__global__ void rsqrt_inplace(float* p, int n) {
  int i = blockIdx.x * TPB + threadIdx.x;
  if (i < n) p[i] = rsqrtf(p[i]);
}

__global__ void edge_norm_kernel(const long long* __restrict__ ei,
                                 const float* __restrict__ dis,
                                 float* __restrict__ nrm, int E) {
  int e = blockIdx.x * TPB + threadIdx.x;
  if (e < E) nrm[e] = dis[(int)ei[e]] * dis[(int)ei[(size_t)E + e]];
}

// ---------------- GCN aggregation (self loop + scatter-add) ----------------

template<int CH>
__global__ void self_init(const float* __restrict__ hw, const float* __restrict__ dis,
                          float* __restrict__ out, int total) {
  int i = blockIdx.x * TPB + threadIdx.x;
  if (i >= total) return;
  int node = i / CH;
  float d = dis[node];
  out[i] = hw[i] * d * d;            // self-loop term: norm = dis[n]^2
}

template<int CH>
__global__ void scatter_edges(const long long* __restrict__ ei,
                              const float* __restrict__ nrm,
                              const float* __restrict__ hw,
                              float* out, int E) {
  int i = blockIdx.x * TPB + threadIdx.x;
  int total = E * CH;                 // <= 102.4M, fits int32
  if (i >= total) return;
  int e = i / CH;
  int c = i % CH;
  int s = (int)ei[e];
  int d = (int)ei[(size_t)E + e];
  atomicAdd(&out[(size_t)d * CH + c], hw[(size_t)s * CH + c] * nrm[e]);
}

template<int CH>
__global__ void bias_relu_kernel(float* buf, const float* __restrict__ b, int total) {
  int i = blockIdx.x * TPB + threadIdx.x;
  if (i >= total) return;
  float v = buf[i] + b[i % CH];
  buf[i] = v > 0.0f ? v : 0.0f;
}

// ---------------- weight pre-pack into WMMA B-fragment order ----------------
// B fragment (16-bit, 32x16): lane l -> column n = l&15; lanes 0-15 carry
// K in {0..7, 16..23}, lanes 16-31 carry K in {8..15, 24..31}; half h -> K = kbase+h (h<8) or kbase+h+8.
// We store per (kstep, ntile, lane) 16 contiguous halves => one 32B load in GEMM.

__global__ void pack_b_kernel(const float* __restrict__ W, _Float16* __restrict__ out,
                              int K, int N) {
  int gid = blockIdx.x * TPB + threadIdx.x;
  int ntiles = N >> 4;
  int total = (K >> 5) * ntiles * 32;
  if (gid >= total) return;
  int lane  = gid & 31;
  int tile  = gid >> 5;
  int kstep = tile / ntiles;
  int ntile = tile % ntiles;
  int n     = ntile * 16 + (lane & 15);
  int kbase = kstep * 32 + ((lane >> 4) << 3);
  _Float16* dst = out + (size_t)gid * 16;
#pragma unroll
  for (int h = 0; h < 16; ++h)
    dst[h] = (_Float16)W[(size_t)(kbase + h + ((h >> 3) << 3)) * N + n];
}

// ---------------- WMMA GEMM: C[M x N] = act(A[M x K] @ W + bias) ----------------
// One wave computes 16 rows x (NT*16) cols. K multiple of 32, N multiple of 16*NT*gridDim.y.
// ACT: 0 = none, 1 = bias+relu, 2 = bias+sigmoid.

template<int NT, int ACT>
__global__ void gemm_wmma_kernel(const float* __restrict__ A,
                                 const _Float16* __restrict__ Bp,
                                 const float* __restrict__ bias,
                                 float* __restrict__ C,
                                 int M, int K, int N) {
  int wave = threadIdx.x >> 5;
  int lane = threadIdx.x & 31;
  int rowtile = blockIdx.x * (TPB / 32) + wave;
  if (rowtile * 16 >= M) return;                 // wave-uniform guard: EXEC stays all-1s
  int row0 = rowtile << 4;
  int coltile0 = blockIdx.y * NT;

  // A fragment lane mapping (16-bit A 16x32): m = lane&15; lane<16 holds
  // K in {0..7, 16..23}, lane>=16 holds K in {8..15, 24..31}.
  int m    = lane & 15;
  int koff = (lane >> 4) << 3;                   // 0 or 8
  const float* arow = A + (size_t)(row0 + m) * K;

  v8f acc[NT];
  v8f zero = {};
#pragma unroll
  for (int t = 0; t < NT; ++t) acc[t] = zero;

  int ntiles = N >> 4;
  for (int ks = 0; ks < K; ks += 32) {
    const float* ap = arow + ks + koff;
    v4f a0 = *(const v4f*)(ap);
    v4f a1 = *(const v4f*)(ap + 4);
    v4f a2 = *(const v4f*)(ap + 16);
    v4f a3 = *(const v4f*)(ap + 20);
    v16h af;
#pragma unroll
    for (int i = 0; i < 4; ++i) {
      af[i]      = (_Float16)a0[i];
      af[4 + i]  = (_Float16)a1[i];
      af[8 + i]  = (_Float16)a2[i];
      af[12 + i] = (_Float16)a3[i];
    }
    const v16h* bp = (const v16h*)Bp + ((size_t)(ks >> 5) * ntiles + coltile0) * 32 + lane;
#pragma unroll
    for (int t = 0; t < NT; ++t) {
      v16h bf = bp[t * 32];
      acc[t] = __builtin_amdgcn_wmma_f32_16x16x32_f16(
          /*neg_a=*/false, af, /*neg_b=*/false, bf,
          /*c_mod=*/(short)0, acc[t], /*reuse_a=*/false, /*reuse_b=*/false);
    }
  }

  // D layout: VGPR r, lanes 0-15 -> M=r, lanes 16-31 -> M=r+8; N = lane&15.
  int rsel = (lane >> 4) << 3;
#pragma unroll
  for (int t = 0; t < NT; ++t) {
    int col = (coltile0 + t) * 16 + (lane & 15);
    float bv = 0.0f;
    if constexpr (ACT != 0) bv = bias[col];
#pragma unroll
    for (int r = 0; r < 8; ++r) {
      float v = acc[t][r];
      if constexpr (ACT != 0) v += bv;           // no dead +0.0f adds for ACT==0
      if constexpr (ACT == 1) v = v > 0.0f ? v : 0.0f;
      if constexpr (ACT == 2) v = 1.0f / (1.0f + __expf(-v));
      C[(size_t)(row0 + rsel + r) * N + col] = v;
    }
  }
}

// ---------------- host-side orchestration ----------------

extern "C" void kernel_launch(void* const* d_in, const int* in_sizes, int n_in,
                              void* d_out, int out_size, void* d_ws, size_t ws_size,
                              hipStream_t stream) {
  const float*     x   = (const float*)d_in[0];
  const long long* ei  = (const long long*)d_in[1];   // int64 edge_index [2, E]
  const float*     W1  = (const float*)d_in[2];
  const float*     b1  = (const float*)d_in[3];
  const float*     W2  = (const float*)d_in[4];
  const float*     b2  = (const float*)d_in[5];
  const float*     Wd1 = (const float*)d_in[6];
  const float*     bd1 = (const float*)d_in[7];
  const float*     Wd2 = (const float*)d_in[8];
  const float*     bd2 = (const float*)d_in[9];
  float* out = (float*)d_out;

  const int IN = 512, H = 64, L = 32;
  const int M = in_sizes[0] / IN;       // 50000 (multiple of 16)
  const int E = in_sizes[1] / 2;        // 1.6M

  // workspace carve-out (256B aligned slices)
  char* wsb = (char*)d_ws;
  size_t off = 0;
  auto carve = [&](size_t bytes) -> void* {
    void* p = wsb + off;
    off = (off + bytes + 255) & ~(size_t)255;
    return p;
  };
  float* dis  = (float*)carve((size_t)M * 4);            // deg -> rsqrt(deg)
  float* nrm  = (float*)carve((size_t)E * 4);
  float* hw1  = (float*)carve((size_t)M * H * 4);
  float* hbuf = (float*)carve((size_t)M * H * 4);
  float* hw2  = (float*)carve((size_t)M * L * 4);
  float* zbuf = (float*)carve((size_t)M * L * 4);
  float* dbuf = (float*)carve((size_t)M * H * 4);
  _Float16* W1p  = (_Float16*)carve((size_t)(IN / 32) * (H / 16) * 512 * 2);
  _Float16* W2p  = (_Float16*)carve((size_t)(H / 32) * (L / 16) * 512 * 2);
  _Float16* Wd1p = (_Float16*)carve((size_t)(L / 32) * (H / 16) * 512 * 2);
  _Float16* Wd2p = (_Float16*)carve((size_t)(H / 32) * (IN / 16) * 512 * 2);

  auto cdiv = [](int a, int b) { return (a + b - 1) / b; };

  // degrees with self loops; dis = rsqrt(deg); per-edge norm
  fill_f32<<<cdiv(M, TPB), TPB, 0, stream>>>(dis, 1.0f, M);
  deg_count<<<cdiv(E, TPB), TPB, 0, stream>>>(ei, dis, E);
  rsqrt_inplace<<<cdiv(M, TPB), TPB, 0, stream>>>(dis, M);
  edge_norm_kernel<<<cdiv(E, TPB), TPB, 0, stream>>>(ei, dis, nrm, E);

  // pre-pack weights into WMMA B-fragment order (f16)
  pack_b_kernel<<<cdiv((IN / 32) * (H / 16) * 32, TPB), TPB, 0, stream>>>(W1, W1p, IN, H);
  pack_b_kernel<<<cdiv((H / 32) * (L / 16) * 32, TPB), TPB, 0, stream>>>(W2, W2p, H, L);
  pack_b_kernel<<<cdiv((L / 32) * (H / 16) * 32, TPB), TPB, 0, stream>>>(Wd1, Wd1p, L, H);
  pack_b_kernel<<<cdiv((H / 32) * (IN / 16) * 32, TPB), TPB, 0, stream>>>(Wd2, Wd2p, H, IN);

  int rowTiles = M / 16;                 // 3125
  int gx = cdiv(rowTiles, TPB / 32);     // 8 waves/block

  // ---- GCN layer 1: hw1 = x @ W1 ; h = relu(agg(hw1) + b1)
  gemm_wmma_kernel<4, 0><<<dim3(gx, 1), TPB, 0, stream>>>(x, W1p, nullptr, hw1, M, IN, H);
  self_init<H><<<cdiv(M * H, TPB), TPB, 0, stream>>>(hw1, dis, hbuf, M * H);
  scatter_edges<H><<<cdiv(E * H, TPB), TPB, 0, stream>>>(ei, nrm, hw1, hbuf, E);
  bias_relu_kernel<H><<<cdiv(M * H, TPB), TPB, 0, stream>>>(hbuf, b1, M * H);

  // ---- GCN layer 2: hw2 = h @ W2 ; z = relu(agg(hw2) + b2)
  gemm_wmma_kernel<2, 0><<<dim3(gx, 1), TPB, 0, stream>>>(hbuf, W2p, nullptr, hw2, M, H, L);
  self_init<L><<<cdiv(M * L, TPB), TPB, 0, stream>>>(hw2, dis, zbuf, M * L);
  scatter_edges<L><<<cdiv(E * L, TPB), TPB, 0, stream>>>(ei, nrm, hw2, zbuf, E);
  bias_relu_kernel<L><<<cdiv(M * L, TPB), TPB, 0, stream>>>(zbuf, b2, M * L);

  // ---- decoder: d = relu(z @ Wd1 + bd1) ; out = sigmoid(d @ Wd2 + bd2)
  gemm_wmma_kernel<4, 1><<<dim3(gx, 1), TPB, 0, stream>>>(zbuf, Wd1p, bd1, dbuf, M, L, H);
  gemm_wmma_kernel<4, 2><<<dim3(gx, IN / 64), TPB, 0, stream>>>(dbuf, Wd2p, bd2, out, M, H, IN);

  (void)n_in; (void)out_size; (void)ws_size;
}
